// SetAbstraction_59966333387153
// MI455X (gfx1250) — matrix operand order repaired and checked by reading, not compiled, not run on hardware
//
#include <hip/hip_runtime.h>
#include <hip/hip_bf16.h>
#include <math.h>

typedef __attribute__((ext_vector_type(16))) _Float16 v16h;
typedef __attribute__((ext_vector_type(8)))  _Float16 v8h;
typedef __attribute__((ext_vector_type(8)))  float    v8f;

#define BB 4
#define CC 128
#define NN 4096
#define MM 2048
#define NB 6
#define INV_SQRT_C 0.0883883476483184f   // 1/sqrt(128)

__device__ __forceinline__ v16h cat16(v8h lo, v8h hi) {
  return __builtin_shufflevector(lo, hi, 0, 1, 2, 3, 4, 5, 6, 7, 8, 9, 10, 11, 12, 13, 14, 15);
}

// ---------------------------------------------------------------------------
// zero accumulators
__global__ void k_init(float* binnum, int* maxnum) {
  int t = threadIdx.x;
  if (t < BB * NB) { binnum[t] = 0.0f; maxnum[t] = 0; }
}

// ---------------------------------------------------------------------------
// G = Wq^T Wk (f16), kkbin[c][j] = sum_o Wq[o,c] * (Wk @ tok)[o,j]
__global__ __launch_bounds__(128) void k_gram(const float* __restrict__ Wq,
                                              const float* __restrict__ Wk,
                                              const float* __restrict__ tok,
                                              _Float16* __restrict__ G16,
                                              float* __restrict__ kkbin) {
  __shared__ float wt[CC * NB];            // (Wk @ tok)[o][j]
  int o = threadIdx.x;
  for (int j = 0; j < NB; ++j) {
    float s = 0.0f;
    for (int c2 = 0; c2 < CC; ++c2) s += Wk[o * CC + c2] * tok[c2 * NB + j];
    wt[o * NB + j] = s;
  }
  __syncthreads();
  int c = threadIdx.x;                     // row of G
  for (int c2 = 0; c2 < CC; ++c2) {
    float s = 0.0f;
    for (int oo = 0; oo < CC; ++oo) s += Wq[oo * CC + c] * Wk[oo * CC + c2];
    G16[c * CC + c2] = (_Float16)s;
  }
  for (int j = 0; j < NB; ++j) {
    float s = 0.0f;
    for (int oo = 0; oo < CC; ++oo) s += Wq[oo * CC + c] * wt[oo * NB + j];
    kkbin[c * NB + j] = s;
  }
}

// ---------------------------------------------------------------------------
// per batch: xbar -> qbar = Wq xbar -> r = Wk^T qbar
__global__ __launch_bounds__(128) void k_vec(const float* __restrict__ x,
                                             const float* __restrict__ Wq,
                                             const float* __restrict__ Wk,
                                             float* __restrict__ rvec) {
  __shared__ float xb[CC], qb[CC];
  int b = blockIdx.x, c = threadIdx.x;
  float s = 0.0f;
  const float* xr = x + ((size_t)(b * CC + c) << 12);
  for (int n = 0; n < NN; ++n) s += xr[n];
  xb[c] = s * (1.0f / NN);
  __syncthreads();
  float q = 0.0f;
  for (int k = 0; k < CC; ++k) q += Wq[c * CC + k] * xb[k];
  qb[c] = q;
  __syncthreads();
  float r = 0.0f;
  for (int o = 0; o < CC; ++o) r += Wk[o * CC + c] * qb[o];
  rvec[b * CC + c] = r;
}

// ---------------------------------------------------------------------------
// y[b,:,m0:m0+16] (f16) = G (128x128 f16) @ x[b,:,m0:m0+16]   (WMMA)
// LDS x-tile stored transposed [m][c] so B-fragments are contiguous b128 loads
__global__ __launch_bounds__(256) void k_kk(const float* __restrict__ x,
                                            const _Float16* __restrict__ G16,
                                            _Float16* __restrict__ y) {
  __shared__ _Float16 ldsG[CC * CC];
  __shared__ _Float16 ldsXT[16 * CC];
  const int b = blockIdx.y, m0 = blockIdx.x << 4;
  const int tid = threadIdx.x, lane = tid & 31, w = tid >> 5;
  {
    uint4* dstv = (uint4*)ldsG;
    const uint4* srcv = (const uint4*)G16;
    for (int t = tid; t < (CC * CC) / 8; t += 256) dstv[t] = srcv[t];
  }
  {
    int c = tid >> 1, mm0 = (tid & 1) << 3;
    const float* src = x + ((size_t)(b * CC + c) << 12) + m0 + mm0;
    #pragma unroll
    for (int i = 0; i < 8; ++i) ldsXT[(mm0 + i) * CC + c] = (_Float16)src[i];
  }
  __syncthreads();
  const int rowM = w * 16 + (lane & 15);
  const int hi8  = (lane >= 16) ? 8 : 0;
  const int hi16 = (lane >= 16) ? 16 : 0;
  const int ml   = lane & 15;
  v8f c8 = {};
  #pragma unroll
  for (int ks = 0; ks < 4; ++ks) {
    const int abase = rowM * CC + ks * 32 + hi8;
    v16h a = cat16(*(const v8h*)&ldsG[abase], *(const v8h*)&ldsG[abase + 16]);
    const int bbase = ml * CC + ks * 32 + hi8;
    v16h bf = cat16(*(const v8h*)&ldsXT[bbase], *(const v8h*)&ldsXT[bbase + 16]);
    c8 = __builtin_amdgcn_wmma_f32_16x16x32_f16(false, a, false, bf, (short)0, c8, false, false);
  }
  #pragma unroll
  for (int r = 0; r < 8; ++r) {
    int row = w * 16 + r + ((lane >= 16) ? 8 : 0);
    y[((size_t)(b * CC + row) << 12) + m0 + ml] = (_Float16)c8[r];
  }
}

// ---------------------------------------------------------------------------
// Flash-style: per 128-row tile, stream all m in 32-col tiles, accumulate
// Z[n]=sum exp(e/sqrt(C)); double-buffered LDS, register-staged global loads.
__global__ __launch_bounds__(256) void k_flash(const float* __restrict__ x,
                                               const _Float16* __restrict__ y,
                                               const float* __restrict__ kkbin,
                                               float* __restrict__ binnum) {
  __shared__ _Float16 ldsA[128 * CC];        // x^T tile, pre-scaled by 1/sqrt(C)
  __shared__ _Float16 ldsBT[2][32 * CC];     // y tile, transposed [m][c], ping-pong
  __shared__ float    ldsbin[CC * NB];
  __shared__ float    ldsZ[128];
  const int b = blockIdx.y, n0 = blockIdx.x << 7;
  const int tid = threadIdx.x, lane = tid & 31, w = tid >> 5;
  for (int t = tid; t < 128 * CC; t += 256) {
    int c = t >> 7, nl = t & 127;
    ldsA[nl * CC + c] = (_Float16)(x[((size_t)(b * CC + c) << 12) + n0 + nl] * INV_SQRT_C);
  }
  for (int t = tid; t < CC * NB; t += 256) ldsbin[t] = kkbin[t];
  __syncthreads();

  const int rowM = w * 16 + (lane & 15);
  const int hi8  = (lane >= 16) ? 8 : 0;
  const int hi16 = (lane >= 16) ? 16 : 0;
  const int ml   = lane & 15;
  v16h a[4];
  #pragma unroll
  for (int ks = 0; ks < 4; ++ks) {
    const int abase = rowM * CC + ks * 32 + hi8;
    a[ks] = cat16(*(const v8h*)&ldsA[abase], *(const v8h*)&ldsA[abase + 16]);
  }

  float zacc[8];
  #pragma unroll
  for (int r = 0; r < 8; ++r) zacc[r] = 0.0f;

  const _Float16* yb = y + ((size_t)b * CC << 12);
  const int c0  = tid >> 1;                  // this thread's y row
  const int mmA = (tid & 1) << 3;            // halves 0..7 or 8..15 of col group A
  const int mmB = mmA + 16;                  // col group B (cols 16..31)

  auto tload = [&](int mt, uint4& sA, uint4& sB) {
    const size_t base = ((size_t)c0 << 12) + (mt << 5);
    sA = *(const uint4*)(yb + base + mmA);
    sB = *(const uint4*)(yb + base + mmB);
  };
  auto tstore = [&](int bufi, uint4 sA, uint4 sB) {
    const _Float16* ha = (const _Float16*)&sA;
    const _Float16* hb = (const _Float16*)&sB;
    #pragma unroll
    for (int i = 0; i < 8; ++i) {
      ldsBT[bufi][(mmA + i) * CC + c0] = ha[i];
      ldsBT[bufi][(mmB + i) * CC + c0] = hb[i];
    }
  };

  {
    uint4 sA, sB;
    tload(0, sA, sB);
    tstore(0, sA, sB);
  }
  __syncthreads();

  const int NT = NN / 32;                    // 128 tiles of 32 cols
  for (int mt = 0; mt < NT; ++mt) {
    const int cur = mt & 1;
    uint4 nA, nB;
    if (mt + 1 < NT) tload(mt + 1, nA, nB);  // issue next tile's global loads early
    #pragma unroll
    for (int s = 0; s < 2; ++s) {
      v8f c8 = {};
      #pragma unroll
      for (int ks = 0; ks < 4; ++ks) {
        const int bbase = (s * 16 + ml) * CC + ks * 32 + hi8;
        v16h bf = cat16(*(const v8h*)&ldsBT[cur][bbase], *(const v8h*)&ldsBT[cur][bbase + 16]);
        c8 = __builtin_amdgcn_wmma_f32_16x16x32_f16(false, a[ks], false, bf, (short)0, c8, false, false);
      }
      #pragma unroll
      for (int r = 0; r < 8; ++r) zacc[r] += __expf(c8[r]);
    }
    if (mt + 1 < NT) tstore(1 ^ cur, nA, nB);
    __syncthreads();
  }

  #pragma unroll
  for (int off = 1; off < 16; off <<= 1)
    #pragma unroll
    for (int r = 0; r < 8; ++r) zacc[r] += __shfl_xor(zacc[r], off, 32);
  if ((lane & 15) == 0) {
    int base = w * 16 + ((lane >= 16) ? 8 : 0);
    #pragma unroll
    for (int r = 0; r < 8; ++r) ldsZ[base + r] = zacc[r];
  }
  __syncthreads();
  if (tid < 128) {
    float Z = ldsZ[tid];
    float eb[NB], tot = Z;
    for (int j = 0; j < NB; ++j) {
      float sb = 0.0f;
      for (int c = 0; c < CC; ++c) sb += (float)ldsA[tid * CC + c] * ldsbin[c * NB + j];
      eb[j] = __expf(sb);                    // ldsA already carries 1/sqrt(C)
      tot += eb[j];
    }
    for (int j = 0; j < NB; ++j) atomicAdd(&binnum[b * NB + j], eb[j] / tot);
  }
}

// ---------------------------------------------------------------------------
// score[b,m] = relu(r_b . x[b,:,m])
__global__ __launch_bounds__(256) void k_score(const float* __restrict__ x,
                                               const float* __restrict__ rvec,
                                               float* __restrict__ score) {
  __shared__ float rl[CC];
  int b = blockIdx.y, m = (blockIdx.x << 8) + threadIdx.x;
  if (threadIdx.x < CC) rl[threadIdx.x] = rvec[b * CC + threadIdx.x];
  __syncthreads();
  float s = 0.0f;
  for (int c = 0; c < CC; ++c) s += rl[c] * x[((size_t)(b * CC + c) << 12) + m];
  score[b * NN + m] = fmaxf(s, 0.0f);
}

__global__ __launch_bounds__(256) void k_stats(const float* __restrict__ score,
                                               float* __restrict__ mu, float* __restrict__ sig) {
  __shared__ float r1[256], r2[256];
  int b = blockIdx.x, t = threadIdx.x;
  float s1 = 0.0f, s2 = 0.0f;
  for (int n = t; n < NN; n += 256) { float v = score[b * NN + n]; s1 += v; s2 += v * v; }
  r1[t] = s1; r2[t] = s2; __syncthreads();
  for (int s = 128; s > 0; s >>= 1) { if (t < s) { r1[t] += r1[t + s]; r2[t] += r2[t + s]; } __syncthreads(); }
  if (t == 0) {
    float m = r1[0] / NN;
    mu[b] = m;
    sig[b] = sqrtf(fmaxf(r2[0] / NN - m * m, 0.0f));
  }
}

// ---------------------------------------------------------------------------
// descending bitonic sort of all B*N z-scores, emit 5 quantile values
__global__ __launch_bounds__(1024) void k_sort(const float* __restrict__ score,
                                               const float* __restrict__ mu,
                                               const float* __restrict__ sig,
                                               float* __restrict__ bvals) {
  __shared__ float d[BB * NN];
  const int n = BB * NN, tid = threadIdx.x;
  for (int t = tid; t < n; t += 1024) {
    int b = t >> 12;
    d[t] = (score[t] - mu[b]) / sig[b] + 1e-8f;
  }
  __syncthreads();
  for (int k = 2; k <= n; k <<= 1)
    for (int jj = k >> 1; jj > 0; jj >>= 1) {
      for (int i = tid; i < n; i += 1024) {
        int ixj = i ^ jj;
        if (ixj > i) {
          bool up = ((i & k) == 0);
          float a = d[i], c = d[ixj];
          bool sw = up ? (a < c) : (a > c);   // descending
          if (sw) { d[i] = c; d[ixj] = a; }
        }
      }
      __syncthreads();
    }
  if (tid < NB - 1) bvals[tid] = d[((tid + 1) * n) / NB];
}

// ---------------------------------------------------------------------------
__global__ __launch_bounds__(256) void k_bin(const float* __restrict__ score,
                                             const float* __restrict__ mu,
                                             const float* __restrict__ sig,
                                             const float* __restrict__ bvals,
                                             int* __restrict__ binidx, int* __restrict__ maxnum) {
  int b = blockIdx.y, nn = (blockIdx.x << 8) + threadIdx.x;
  float s = (score[b * NN + nn] - mu[b]) / sig[b] + 1e-8f;
  int j = 0;
  for (int k = 0; k < NB - 1; ++k) j += (s < bvals[k]) ? 1 : 0;
  binidx[b * NN + nn] = j;
  atomicAdd(&maxnum[b * NB + j], 1);
}

// ---------------------------------------------------------------------------
// capped renormalization -> chosen counts + prefix offsets (rows sum to M)
__global__ void k_choose(const float* __restrict__ binnum, const int* __restrict__ maxnum,
                         int* __restrict__ chosen, int* __restrict__ offs) {
  int b = threadIdx.x;
  if (b >= BB) return;
  float bn[NB], tot = 0.0f, mx[NB], p[NB], ch[NB];
  for (int j = 0; j < NB; ++j) { bn[j] = binnum[b * NB + j]; tot += bn[j]; }
  for (int j = 0; j < NB; ++j) {
    mx[j] = (float)maxnum[b * NB + j];
    p[j] = (bn[j] / tot) * mx[j] + 1e-10f;
    ch[j] = 0.0f;
  }
  for (int it = 0; it < NB; ++it) {
    float ps = 0.0f, cs = 0.0f;
    for (int j = 0; j < NB; ++j) { ps += p[j]; cs += ch[j]; }
    ps = fmaxf(ps, 1e-30f);
    float need = (float)MM - cs;
    for (int j = 0; j < NB; ++j) {
      p[j] /= ps;
      ch[j] = fminf(ch[j] + p[j] * need, mx[j]);
      if (ch[j] >= mx[j]) p[j] = 0.0f;
    }
  }
  int chi[NB], sum = 0;
  for (int j = 0; j < NB; ++j) { chi[j] = (int)ch[j]; sum += chi[j]; }
  int rem = MM - sum, best = 0, bv = maxnum[b * NB] - chi[0];
  for (int j = 1; j < NB; ++j) {
    int v = maxnum[b * NB + j] - chi[j];
    if (v > bv) { bv = v; best = j; }
  }
  chi[best] += rem;
  int off = 0;
  for (int j = 0; j < NB; ++j) { offs[b * NB + j] = off; chosen[b * NB + j] = chi[j]; off += chi[j]; }
}

// ---------------------------------------------------------------------------
// per (b,bin): Gumbel keys, descending bitonic sort w/ index payload,
// emit first chosen[b,j] indices at prefix offset
__global__ __launch_bounds__(512) void k_topk(const float* __restrict__ score,
                                              const float* __restrict__ mu,
                                              const float* __restrict__ sig,
                                              const int* __restrict__ binidx,
                                              const float* __restrict__ noise,
                                              const int* __restrict__ chosen,
                                              const int* __restrict__ offs,
                                              int* __restrict__ idxbuf) {
  __shared__ float key[NN];
  __shared__ int   sidx[NN];
  __shared__ float red[512];
  const int b = blockIdx.x / NB, j = blockIdx.x % NB, tid = threadIdx.x;
  float part = 0.0f;
  for (int n = tid; n < NN; n += 512) {
    float z = (score[b * NN + n] - mu[b]) / sig[b];
    float e = (binidx[b * NN + n] == j) ? __expf(tanhf(z) * 10.0f) : 0.0f;
    key[n] = e; sidx[n] = n; part += e;
  }
  red[tid] = part; __syncthreads();
  for (int s = 256; s > 0; s >>= 1) { if (tid < s) red[tid] += red[tid + s]; __syncthreads(); }
  const float S = red[0];
  __syncthreads();
  for (int n = tid; n < NN; n += 512) {
    float sp = (S > 0.0f) ? key[n] / S : 1e-8f;
    float u = noise[(size_t)(b * NB + j) * NN + n];
    u = fminf(fmaxf(u, 1e-12f), 1.0f - 1e-12f);
    key[n] = logf(sp + 1e-30f) - logf(-logf(u));
  }
  __syncthreads();
  for (int k = 2; k <= NN; k <<= 1)
    for (int jj = k >> 1; jj > 0; jj >>= 1) {
      for (int i = tid; i < NN; i += 512) {
        int ixj = i ^ jj;
        if (ixj > i) {
          bool up = ((i & k) == 0);
          float a = key[i], c = key[ixj];
          bool sw = up ? (a < c) : (a > c);
          if (sw) {
            key[i] = c; key[ixj] = a;
            int t1 = sidx[i]; sidx[i] = sidx[ixj]; sidx[ixj] = t1;
          }
        }
      }
      __syncthreads();
    }
  const int cnt = chosen[b * NB + j], off = offs[b * NB + j];
  for (int t = tid; t < cnt; t += 512) idxbuf[b * MM + off + t] = sidx[t];
}

// ---------------------------------------------------------------------------
__global__ __launch_bounds__(256) void k_gather(const float* __restrict__ x,
                                                const int* __restrict__ idxbuf,
                                                float* __restrict__ out) {
  int t = (blockIdx.x << 8) + threadIdx.x;      // over B*C*M
  int m = t & (MM - 1);
  int bc = t >> 11;                             // b*128 + c
  int b = t >> 18;
  out[t] = x[((size_t)bc << 12) + idxbuf[(b << 11) + m]];
}

__global__ __launch_bounds__(256) void k_outidx(const int* __restrict__ idxbuf,
                                                float* __restrict__ out) {
  int t = (blockIdx.x << 8) + threadIdx.x;      // over B*M
  out[BB * CC * MM + t] = (float)idxbuf[t];
}

// ---------------------------------------------------------------------------
extern "C" void kernel_launch(void* const* d_in, const int* in_sizes, int n_in,
                              void* d_out, int out_size, void* d_ws, size_t ws_size,
                              hipStream_t stream) {
  const float* x    = (const float*)d_in[0];
  const float* Wq   = (const float*)d_in[1];
  const float* Wk   = (const float*)d_in[2];
  const float* tok  = (const float*)d_in[3];
  const float* nois = (const float*)d_in[4];
  float* out = (float*)d_out;

  char* w = (char*)d_ws;
  size_t off = 0;
  auto carve = [&](size_t bytes) { void* p = w + off; off = (off + bytes + 255) & ~(size_t)255; return p; };
  _Float16* G16   = (_Float16*)carve((size_t)CC * CC * 2);
  float*    kkbin = (float*)   carve((size_t)CC * NB * 4);
  float*    rvec  = (float*)   carve((size_t)BB * CC * 4);
  float*    score = (float*)   carve((size_t)BB * NN * 4);
  float*    mu    = (float*)   carve((size_t)BB * 4);
  float*    sig   = (float*)   carve((size_t)BB * 4);
  _Float16* y16   = (_Float16*)carve((size_t)BB * CC * NN * 2);
  float*    binnm = (float*)   carve((size_t)BB * NB * 4);
  int*      maxnm = (int*)     carve((size_t)BB * NB * 4);
  int*      binix = (int*)     carve((size_t)BB * NN * 4);
  float*    bvals = (float*)   carve((size_t)(NB - 1) * 4);
  int*      chos  = (int*)     carve((size_t)BB * NB * 4);
  int*      offs  = (int*)     carve((size_t)BB * NB * 4);
  int*      idxb  = (int*)     carve((size_t)BB * MM * 4);

  k_init  <<<1, 64, 0, stream>>>(binnm, maxnm);
  k_gram  <<<1, 128, 0, stream>>>(Wq, Wk, tok, G16, kkbin);
  k_vec   <<<BB, 128, 0, stream>>>(x, Wq, Wk, rvec);
  k_kk    <<<dim3(NN / 16, BB), 256, 0, stream>>>(x, G16, y16);
  k_score <<<dim3(NN / 256, BB), 256, 0, stream>>>(x, rvec, score);
  k_stats <<<BB, 256, 0, stream>>>(score, mu, sig);
  k_flash <<<dim3(NN / 128, BB), 256, 0, stream>>>(x, y16, kkbin, binnm);
  k_sort  <<<1, 1024, 0, stream>>>(score, mu, sig, bvals);
  k_bin   <<<dim3(NN / 256, BB), 256, 0, stream>>>(score, mu, sig, bvals, binix, maxnm);
  k_choose<<<1, 32, 0, stream>>>(binnm, maxnm, chos, offs);
  k_topk  <<<BB * NB, 512, 0, stream>>>(score, mu, sig, binix, nois, chos, offs, idxb);
  k_gather<<<(BB * CC * MM) / 256, 256, 0, stream>>>(x, idxb, out);
  k_outidx<<<(BB * MM) / 256, 256, 0, stream>>>(idxb, out);
}
